// RNN_2826088480793
// MI455X (gfx1250) — matrix operand order repaired and checked by reading, not compiled, not run on hardware
//
#include <hip/hip_runtime.h>
#include <math.h>

// ---------------------------------------------------------------------------
// Stacked tanh-RNN (T=4096, B=128, D=64; 20xH20 + 10xH10 + 1xH1 layers).
// MI455X strategy:
//  * Per layer: parallel WMMA GEMM for the hoisted input projection
//    (xw = Y @ Wih^T + bih) over M = T*B rows, V_WMMA_F32_16X16X32_BF16.
//  * Sequential scan: one workgroup, 8 wave32s x 16 batch rows each; hidden
//    tile kept in registers, one WMMA per 16-col tile per step; D->A operand
//    relayout via a tiny per-wave LDS bounce.  Batch rows are independent so
//    the T-loop has no barriers.
//  * All inter-kernel tensors are stored in WMMA *operand lane order*:
//      xw  : C-layout  [mtile][nt][lane] x 8 f32  (32B contig per lane)
//      y   : A-layout  [mtile][lane]     x 16 bf16 (32B contig per lane)
//    so the serial scan does 2x b128 loads + 2x b128 stores per step/wave
//    (single cacheline each) instead of scattered dwords.
//  * Working set (xw 64MB + 2x32MB activation ping-pong) is L2-resident
//    (192MB); only x (134MB) streams from HBM, once.
//  * Weights pre-packed once into bf16 WMMA B-operand layout, zero-padded,
//    so padding columns compute to exactly zero (no masking in hot loops).
// ---------------------------------------------------------------------------

typedef __attribute__((ext_vector_type(16))) __bf16          v16bf;
typedef __attribute__((ext_vector_type(16))) unsigned short  v16u;
typedef __attribute__((ext_vector_type(8)))  unsigned short  v8u;
typedef __attribute__((ext_vector_type(8)))  float           v8f;
typedef __attribute__((ext_vector_type(8)))  float           f8;

static constexpr int T_ = 4096, B_ = 128, M_ = T_ * B_;
static constexpr int MT_ = M_ / 16;   // 32768 M-tiles
static constexpr int SLOT = 2048;     // ushorts per packed-weight slot (4KB)

union BCast { v16u u; v16bf b; };

__device__ __forceinline__ unsigned short f2bf(float f) {
  return __builtin_bit_cast(unsigned short, (__bf16)f);  // native cvt on gfx1250
}

__device__ __forceinline__ v8f wmma_bf16(v16u a, v16u b, v8f c) {
  BCast A; A.u = a; BCast Bm; Bm.u = b;
  return __builtin_amdgcn_wmma_f32_16x16x32_bf16(false, A.b, false, Bm.b,
                                                 (short)0, c, false, false);
}

// ---------------------------------------------------------------------------
// Pack one f32 weight matrix (hout x hin, row-major) into bf16 WMMA B-operand
// layout tiles (32x16), zero-padded.  Tile index = kt*NT + nt, 1KB per tile.
// B layout (16-bit, 32x16): lane<16 -> N=lane, K {0..7,16..23} packed 2/dword;
//                           lane>=16 -> N=lane-16, K {8..15,24..31}.
// ---------------------------------------------------------------------------
__global__ void pack_w(const float* __restrict__ w, int hout, int hin,
                       long mstride, unsigned short* __restrict__ wpack,
                       int slot0) {
  const float* wm = w + (long)blockIdx.x * mstride;
  const int tile = threadIdx.x >> 5, lane = threadIdx.x & 31;
  const int ktc = (hin + 31) >> 5, ntc = (hout + 15) >> 4;
  if (tile >= ktc * ntc) return;
  const int nt = tile % ntc;
  const int kt = tile / ntc;
  const int n  = nt * 16 + (lane & 15);
  const int kb = kt * 32 + ((lane >> 4) << 3);
  unsigned short* out =
      wpack + (long)(slot0 + blockIdx.x) * SLOT + tile * 512 + lane * 16;
  for (int e = 0; e < 16; ++e) {
    int k = kb + e + ((e >> 3) << 3);
    out[e] = (n < hout && k < hin) ? f2bf(wm[(long)n * hin + k])
                                   : (unsigned short)0;
  }
}

// ---------------------------------------------------------------------------
// Input-projection GEMM: xw = A @ W^T + bih.  8 waves/block, one 16-row
// M-tile per wave, K-chained WMMAs.  Activation input arrives already in
// A-operand lane order (32B/lane); xw leaves in C-operand lane order.
// A layout (16-bit, 16x32): lane<16 -> M=lane, K {0..7,16..23};
//                           lane>=16 -> M=lane-16, K {8..15,24..31}.
// ---------------------------------------------------------------------------
template <int KT, int NT, bool F32IN>
__global__ __launch_bounds__(256)
void rnn_gemm(const float* __restrict__ a32,
              const unsigned short* __restrict__ a16,
              const unsigned short* __restrict__ wslot, int hout,
              const float* __restrict__ bih, float* __restrict__ xw) {
  const int lane = threadIdx.x & 31, wave = threadIdx.x >> 5;
  const int lhi = lane >> 4, lrow = lane & 15;
  const long mtile = (long)blockIdx.x * 8 + wave;

  v16u bt[KT * NT];
#pragma unroll
  for (int tI = 0; tI < KT * NT; ++tI) {
    const v8u* p = (const v8u*)(wslot + tI * 512 + lane * 16);
    v8u lo = p[0], hi = p[1];
#pragma unroll
    for (int i = 0; i < 8; ++i) { bt[tI][i] = lo[i]; bt[tI][i + 8] = hi[i]; }
  }

  v8f acc[NT];
#pragma unroll
  for (int nt = 0; nt < NT; ++nt)
#pragma unroll
    for (int i = 0; i < 8; ++i) acc[nt][i] = 0.f;

#pragma unroll
  for (int kt = 0; kt < KT; ++kt) {
    v16u av;
    if constexpr (F32IN) {     // raw x input: f32 row-major, row stride 64
      const long row = mtile * 16 + lrow;
      const int koff = kt * 32 + lhi * 8;
      const f8* ap = (const f8*)(a32 + row * 64 + koff);
      f8 f0 = ap[0], f1 = ap[2];
#pragma unroll
      for (int i = 0; i < 8; ++i) { av[i] = f2bf(f0[i]); av[i + 8] = f2bf(f1[i]); }
    } else {                   // activation input: bf16 in A-operand order
      av = *(const v16u*)(a16 + (mtile * 32 + lane) * 16);
    }
#pragma unroll
    for (int nt = 0; nt < NT; ++nt)
      acc[nt] = wmma_bf16(av, bt[kt * NT + nt], acc[nt]);
  }

#pragma unroll
  for (int nt = 0; nt < NT; ++nt) {
    const int n = nt * 16 + lrow;
    const float b = (n < hout) ? bih[n] : 0.f;
    v8f o;
#pragma unroll
    for (int i = 0; i < 8; ++i) o[i] = acc[nt][i] + b;  // pad cols stay 0
    *(v8f*)(xw + ((mtile * NT + nt) * 32 + lane) * 8) = o;
  }
}

// ---------------------------------------------------------------------------
// Sequential scan: h_t = tanh(xw_t + h_{t-1} @ Whh^T + bhh).
// One block, 8 waves x 16 batch rows; hidden tile in registers (A layout);
// per-wave private 1KB LDS region for the C->A relayout each step.
// xw read in C-operand order; y written in A-operand order.
// ---------------------------------------------------------------------------
template <int NT>
__global__ __launch_bounds__(256)
void rnn_scan(const float* __restrict__ xw,
              const unsigned short* __restrict__ wslot,
              const float* __restrict__ bhh, int hout,
              unsigned short* __restrict__ yout,
              float* __restrict__ finalY, float* __restrict__ finalH) {
  __shared__ unsigned short lds[8 * 16 * 32];
  for (int i = threadIdx.x; i < 8 * 16 * 32; i += 256) lds[i] = 0;
  __syncthreads();

  const int lane = threadIdx.x & 31, wave = threadIdx.x >> 5;
  const int lhi = lane >> 4, lrow = lane & 15;
  const int rbase = wave * 16 + lhi * 8;

  v16u wb[NT];
  float bb[NT];
#pragma unroll
  for (int nt = 0; nt < NT; ++nt) {
    const v8u* p = (const v8u*)(wslot + nt * 512 + lane * 16);
    v8u lo = p[0], hi = p[1];
#pragma unroll
    for (int i = 0; i < 8; ++i) { wb[nt][i] = lo[i]; wb[nt][i + 8] = hi[i]; }
    const int n = nt * 16 + lrow;
    bb[nt] = (n < hout) ? bhh[n] : 0.f;
  }

  v16u hA;                            // hidden state, A-operand layout, h0 = 0
#pragma unroll
  for (int i = 0; i < 16; ++i) hA[i] = 0;

  unsigned short* myLds = lds + wave * 512;

  for (int t = 0; t < T_; ++t) {
    const long mtile = (long)t * 8 + wave;
    __builtin_prefetch(xw + (((mtile + 64) * NT) * 32 + lane) * 8, 0, 3);
#pragma unroll
    for (int nt = 0; nt < NT; ++nt) {
      const int col = nt * 16 + lrow;
      v8f c = *(const v8f*)(xw + ((mtile * NT + nt) * 32 + lane) * 8);
      v8f acc = wmma_bf16(hA, wb[nt], c);       // C carries xw + bih
      float hv[8];
#pragma unroll
      for (int i = 0; i < 8; ++i) hv[i] = tanhf(acc[i] + bb[nt]);
#pragma unroll
      for (int i = 0; i < 8; ++i)
        myLds[(lhi * 8 + i) * 32 + col] = f2bf(hv[i]);              // relayout
      if (finalY && nt == 0 && lrow == 0) {     // H=1 output column (lanes 0,16)
#pragma unroll
        for (int i = 0; i < 8; ++i) finalY[(long)t * B_ + rbase + i] = hv[i];
        if (t == T_ - 1)
#pragma unroll
          for (int i = 0; i < 8; ++i) finalH[rbase + i] = hv[i];
      }
    }
    // rebuild A-layout hidden operand from LDS (two ds_load_b128 per lane)
    const unsigned short* rp = myLds + lrow * 32 + lhi * 8;
    v8u lo = *(const v8u*)rp;
    v8u hi = *(const v8u*)(rp + 16);
#pragma unroll
    for (int i = 0; i < 8; ++i) { hA[i] = lo[i]; hA[i + 8] = hi[i]; }
    // layer output, already in A-operand order for the next layer's GEMM
    *(v16u*)(yout + (mtile * 32 + lane) * 16) = hA;
  }
}

// ---------------------------------------------------------------------------
extern "C" void kernel_launch(void* const* d_in, const int* in_sizes, int n_in,
                              void* d_out, int out_size, void* d_ws,
                              size_t ws_size, hipStream_t stream) {
  (void)in_sizes; (void)n_in; (void)out_size; (void)ws_size;
  const float* x       = (const float*)d_in[0];
  const float* s1_wih0 = (const float*)d_in[1];
  const float* s1_whh0 = (const float*)d_in[2];
  const float* s1_bih0 = (const float*)d_in[3];
  const float* s1_bhh0 = (const float*)d_in[4];
  const float* s1_wih  = (const float*)d_in[5];
  const float* s1_whh  = (const float*)d_in[6];
  const float* s1_bih  = (const float*)d_in[7];
  const float* s1_bhh  = (const float*)d_in[8];
  const float* s2_wih0 = (const float*)d_in[9];
  const float* s2_whh0 = (const float*)d_in[10];
  const float* s2_bih0 = (const float*)d_in[11];
  const float* s2_bhh0 = (const float*)d_in[12];
  const float* s2_wih  = (const float*)d_in[13];
  const float* s2_whh  = (const float*)d_in[14];
  const float* s2_bih  = (const float*)d_in[15];
  const float* s2_bhh  = (const float*)d_in[16];
  const float* s3_wih  = (const float*)d_in[17];
  const float* s3_whh  = (const float*)d_in[18];
  const float* s3_bih  = (const float*)d_in[19];
  const float* s3_bhh  = (const float*)d_in[20];

  // workspace: [62 x 4KB packed weights][xw f32 64MB][yA 32MB][yB 32MB]
  char* ws = (char*)d_ws;
  unsigned short* wpack = (unsigned short*)ws;
  size_t off = 62 * 4096;
  float* xwp = (float*)(ws + off);                  off += (size_t)MT_ * 2 * 32 * 8 * 4;
  unsigned short* yA = (unsigned short*)(ws + off); off += (size_t)MT_ * 32 * 16 * 2;
  unsigned short* yB = (unsigned short*)(ws + off);

  // --- pack all weights into WMMA B-operand layout (slots) ---
  pack_w<<<1,  128, 0, stream>>>(s1_wih0, 20, 64,   0, wpack,  0);
  pack_w<<<1,  128, 0, stream>>>(s1_whh0, 20, 20,   0, wpack,  1);
  pack_w<<<19, 128, 0, stream>>>(s1_wih,  20, 20, 400, wpack,  2);
  pack_w<<<19, 128, 0, stream>>>(s1_whh,  20, 20, 400, wpack, 21);
  pack_w<<<1,  128, 0, stream>>>(s2_wih0, 10, 20,   0, wpack, 40);
  pack_w<<<1,  128, 0, stream>>>(s2_whh0, 10, 10,   0, wpack, 41);
  pack_w<<<9,  128, 0, stream>>>(s2_wih,  10, 10, 100, wpack, 42);
  pack_w<<<9,  128, 0, stream>>>(s2_whh,  10, 10, 100, wpack, 51);
  pack_w<<<1,  128, 0, stream>>>(s3_wih,   1, 10,   0, wpack, 60);
  pack_w<<<1,  128, 0, stream>>>(s3_whh,   1,  1,   0, wpack, 61);

  const int GB = MT_ / 8;   // 4096 blocks, 8 M-tiles each

  // --- stack 1 ---
  rnn_gemm<2, 2, true><<<GB, 256, 0, stream>>>(x, nullptr, wpack + 0 * SLOT,
                                               20, s1_bih0, xwp);
  rnn_scan<2><<<1, 256, 0, stream>>>(xwp, wpack + 1 * SLOT, s1_bhh0, 20, yA,
                                     nullptr, nullptr);
  unsigned short* cur = yA;
  unsigned short* nxt = yB;
  for (int l = 0; l < 19; ++l) {
    rnn_gemm<1, 2, false><<<GB, 256, 0, stream>>>(
        nullptr, cur, wpack + (size_t)(2 + l) * SLOT, 20, s1_bih + l * 20, xwp);
    rnn_scan<2><<<1, 256, 0, stream>>>(
        xwp, wpack + (size_t)(21 + l) * SLOT, s1_bhh + l * 20, 20, nxt,
        nullptr, nullptr);
    unsigned short* t = cur; cur = nxt; nxt = t;
  }

  // --- stack 2 ---
  rnn_gemm<1, 1, false><<<GB, 256, 0, stream>>>(nullptr, cur, wpack + 40 * SLOT,
                                                10, s2_bih0, xwp);
  rnn_scan<1><<<1, 256, 0, stream>>>(xwp, wpack + 41 * SLOT, s2_bhh0, 10, nxt,
                                     nullptr, nullptr);
  { unsigned short* t = cur; cur = nxt; nxt = t; }
  for (int l = 0; l < 9; ++l) {
    rnn_gemm<1, 1, false><<<GB, 256, 0, stream>>>(
        nullptr, cur, wpack + (size_t)(42 + l) * SLOT, 10, s2_bih + l * 10, xwp);
    rnn_scan<1><<<1, 256, 0, stream>>>(
        xwp, wpack + (size_t)(51 + l) * SLOT, s2_bhh + l * 10, 10, nxt,
        nullptr, nullptr);
    unsigned short* t = cur; cur = nxt; nxt = t;
  }

  // --- stack 3 (final; writes f32 y (T,B,1) + hT (1,B,1) to d_out) ---
  rnn_gemm<1, 1, false><<<GB, 256, 0, stream>>>(nullptr, cur, wpack + 60 * SLOT,
                                                1, s3_bih, xwp);
  rnn_scan<1><<<1, 256, 0, stream>>>(xwp, wpack + 61 * SLOT, s3_bhh, 1, nxt,
                                     (float*)d_out, (float*)d_out + (size_t)M_);
}
